// CustomLSTMClassifier_49039936585927
// MI455X (gfx1250) — matrix operand order repaired and checked by reading, not compiled, not run on hardware
//
#include <hip/hip_runtime.h>
#include <hip/hip_bf16.h>

typedef __attribute__((ext_vector_type(16))) __bf16 v16bf;
typedef __attribute__((ext_vector_type(8)))  __bf16 v8bf;
typedef __attribute__((ext_vector_type(8)))  float  v8f;

#define B_   128
#define T_   512
#define I_   256
#define H_   512
#define G4H  2048
#define C_   4

// ---------------- math helpers ----------------
__device__ __forceinline__ float fast_sigmoid(float x) {
    return 1.0f / (1.0f + __expf(-x));
}
__device__ __forceinline__ float fast_tanh(float x) {
    float t = __expf(-2.0f * fabsf(x));     // in (0,1]
    float r = (1.0f - t) / (1.0f + t);
    return copysignf(r, x);
}

// ---------------- CDNA5 async global->LDS (16B granule), via inline asm ----------------
// global_load_async_to_lds_b128 vdst, vaddr, off :
//   LDS[vdst(32-bit lds byte addr)] = MEM[vaddr(64-bit)] , tracked by ASYNCcnt.
// Low 32 bits of a generic shared pointer are the allocation-relative LDS address
// (ISA: flat->LDS aperture uses addr[31:0]).
__device__ __forceinline__ void async_copy16(const void* gsrc, void* lds_dst) {
    unsigned ldsa = (unsigned)(unsigned long long)(__SIZE_TYPE__)lds_dst;
    unsigned long long ga = (unsigned long long)(__SIZE_TYPE__)gsrc;
    asm volatile("global_load_async_to_lds_b128 %0, %1, off"
                 :: "v"(ldsa), "v"(ga)
                 : "memory");
}
__device__ __forceinline__ void wait_async0() {
#if __has_builtin(__builtin_amdgcn_s_wait_asynccnt)
    __builtin_amdgcn_s_wait_asynccnt(0);
#else
    asm volatile("s_wait_asynccnt 0x0" ::: "memory");
#endif
}

// ---------------- grid-wide barrier (persistent kernel) ----------------
__device__ __forceinline__ void grid_sync(unsigned* cnt, unsigned* gen, unsigned nb) {
    __syncthreads();
    if (threadIdx.x == 0) {
        __threadfence();
        unsigned g = __hip_atomic_load(gen, __ATOMIC_RELAXED, __HIP_MEMORY_SCOPE_AGENT);
        unsigned arrived = __hip_atomic_fetch_add(cnt, 1u, __ATOMIC_ACQ_REL, __HIP_MEMORY_SCOPE_AGENT);
        if (arrived == nb - 1u) {
            __hip_atomic_store(cnt, 0u, __ATOMIC_RELAXED, __HIP_MEMORY_SCOPE_AGENT);
            __hip_atomic_store(gen, g + 1u, __ATOMIC_RELEASE, __HIP_MEMORY_SCOPE_AGENT);
        } else {
            while (__hip_atomic_load(gen, __ATOMIC_ACQUIRE, __HIP_MEMORY_SCOPE_AGENT) == g) { }
        }
    }
    __syncthreads();
}

// ---------------- prep kernels ----------------
__global__ void f32_to_bf16_kernel(const float* __restrict__ in,
                                   __hip_bfloat16* __restrict__ out, int n) {
    int i = blockIdx.x * blockDim.x + threadIdx.x;
    if (i < n) out[i] = __float2bfloat16(in[i]);
}

// W [K,N] f32 row-major  ->  Wt [N,K] bf16 (so WMMA B-fragments are contiguous)
__global__ void transpose_to_bf16_kernel(const float* __restrict__ W,
                                         __hip_bfloat16* __restrict__ Wt,
                                         int K, int N) {
    int i = blockIdx.x * blockDim.x + threadIdx.x;
    if (i < K * N) {
        int n = i / K, k = i - n * K;
        Wt[i] = __float2bfloat16(W[(size_t)k * N + n]);
    }
}

__global__ void init_state_kernel(__hip_bfloat16* __restrict__ h,
                                  float* __restrict__ c,
                                  unsigned* __restrict__ bar) {
    int i = blockIdx.x * blockDim.x + threadIdx.x;
    if (i < B_ * H_) { h[i] = __float2bfloat16(0.0f); c[i] = 0.0f; }
    if (i < 2) bar[i] = 0u;
}

// ---------------- big input-projection GEMM (bf16 WMMA, f32 accum) ----------------
// Out[M,N] = A(bf16)[M,K] @ Wt(bf16,[N,K])^T + bias[N]
// modeInterleaved=1: row m maps to A + (b*T + t)*K with b=m&127, t=m>>7   (x layout [B,T,I])
// modeInterleaved=0: row m maps to A + m*K                                (h1_seq layout [T*B,H])
__global__ __launch_bounds__(256) void wmma_gemm_bf16(
    const __hip_bfloat16* __restrict__ A,
    const __hip_bfloat16* __restrict__ Wt,
    const float* __restrict__ bias,
    float* __restrict__ Out,
    int M, int N, int K, int modeInterleaved)
{
    const int lane = threadIdx.x & 31;
    const int wave = threadIdx.x >> 5;
    const int gw   = blockIdx.x * 8 + wave;
    const int hi   = lane >> 4;          // 0 or 1
    const int ln   = lane & 15;
    const int nTN  = N >> 4;
    const int ntiles = (M >> 4) * nTN;

    for (int tile = gw; tile < ntiles; tile += gridDim.x * 8) {
        const int mt = tile / nTN;
        const int nt = tile - mt * nTN;
        const int m  = mt * 16 + ln;

        const __hip_bfloat16* arow;
        if (modeInterleaved) {
            int b = m & (B_ - 1), t = m >> 7;
            arow = A + ((size_t)b * T_ + t) * (size_t)K;
        } else {
            arow = A + (size_t)m * K;
        }
        const __hip_bfloat16* brow = Wt + (size_t)(nt * 16 + ln) * K + hi * 16;

        v8f acc = {};
        for (int k0 = 0; k0 < K; k0 += 32) {
            const int kb = k0 + hi * 16;
            v8bf alo = *(const v8bf*)(arow + kb);
            v8bf ahi = *(const v8bf*)(arow + kb + 8);
            v16bf af = __builtin_shufflevector(alo, ahi,
                        0,1,2,3,4,5,6,7,8,9,10,11,12,13,14,15);
            v16bf wf = *(const v16bf*)(brow + k0);
            acc = __builtin_amdgcn_wmma_f32_16x16x32_bf16(
                      false, af, false, wf, (short)0, acc, false, false);
        }

        const int coln = nt * 16 + ln;
        const float bv = bias[coln];
#pragma unroll
        for (int r = 0; r < 8; ++r) {
            const int row = mt * 16 + r + hi * 8;
            Out[(size_t)row * N + coln] = acc[r] + bv;
        }
    }
}

// ---------------- persistent LSTM layer (recurrence over T in one kernel) ----------------
// gates[B,4H] = xp[t] + h @ W_h + b_h ; then elementwise LSTM update.
// Launch with EXACTLY 128 blocks x 256 threads (1024 waves == 1024 output tiles)
// and 147456 bytes of dynamic LDS (16KB h-tile + 8 waves x 16KB W_h cache).
__global__ __launch_bounds__(256) void lstm_layer_persistent(
    const float* __restrict__ xp,            // [T*B, 4H] f32 (input proj + b_x)
    const __hip_bfloat16* __restrict__ WhT,  // [4H, H] bf16 (transposed W_h)
    const float* __restrict__ bh,            // [4H]
    __hip_bfloat16* __restrict__ h,          // [B, H]  state (bf16)
    float* __restrict__ c,                   // [B, H]  state (f32)
    float* __restrict__ gates,               // [B, 4H] scratch (f32)
    __hip_bfloat16* __restrict__ h_seq,      // [T*B, H] or nullptr
    unsigned* __restrict__ bar)              // [2]: count, generation
{
    extern __shared__ __hip_bfloat16 lds[];  // 8192 (h tile) + 8*8192 (W) halfs
    const int lane = threadIdx.x & 31;
    const int wave = threadIdx.x >> 5;
    const int gw   = blockIdx.x * 8 + wave;  // 0..1023
    const int hi   = lane >> 4;
    const int ln   = lane & 15;
    const int mt   = gw >> 7;                // 0..7   (B tiles) -- uniform per block
    const int nt   = gw & 127;               // 0..127 (4H tiles)

    __hip_bfloat16* htile = lds;                               // 16 x 512 bf16
    __hip_bfloat16* myLds = lds + 8192 + (size_t)wave * 8192;  // W_h fragments

    // ---- preload this wave's 16x512 slice of WhT into LDS (async global->LDS) ----
    const __hip_bfloat16* wrow = WhT + (size_t)(nt * 16 + ln) * H_ + hi * 16;
#pragma unroll 4
    for (int kk = 0; kk < 16; ++kk) {
        async_copy16(wrow + kk * 32,     myLds + kk * 512 + lane * 16);
        async_copy16(wrow + kk * 32 + 8, myLds + kk * 512 + lane * 16 + 8);
    }
    wait_async0();
    __syncthreads();

    const int   coln = nt * 16 + ln;
    const float bhv  = bh[coln];
    const unsigned nb = gridDim.x;
    const int tid  = blockIdx.x * 256 + threadIdx.x;
    const int nthr = (int)nb * 256;

    const char* hsrc = (const char*)(h + (size_t)mt * 16 * H_);  // 16KB contiguous
    const int   cbase = (int)threadIdx.x * 64;                   // 64B per thread

    for (int t = 0; t < T_; ++t) {
        // ---- stage this block's 16 rows of h into LDS via async copy ----
#pragma unroll
        for (int q = 0; q < 4; ++q)
            async_copy16(hsrc + cbase + q * 16, (char*)htile + cbase + q * 16);
        wait_async0();
        __syncthreads();

        // ---- Phase A: gates tile = h @ W_h (WMMA) + xp[t] + b_h ----
        v8f acc = {};
        const __hip_bfloat16* arow = htile + (size_t)ln * H_;
#pragma unroll 4
        for (int kk = 0; kk < 16; ++kk) {
            const int kb = kk * 32 + hi * 16;
            v8bf alo = *(const v8bf*)(arow + kb);
            v8bf ahi = *(const v8bf*)(arow + kb + 8);
            v16bf af = __builtin_shufflevector(alo, ahi,
                        0,1,2,3,4,5,6,7,8,9,10,11,12,13,14,15);
            v16bf wf = *(const v16bf*)(myLds + kk * 512 + lane * 16);
            acc = __builtin_amdgcn_wmma_f32_16x16x32_bf16(
                      false, af, false, wf, (short)0, acc, false, false);
        }
        const float* xprow = xp + (size_t)t * B_ * G4H;
#pragma unroll
        for (int r = 0; r < 8; ++r) {
            const int row = mt * 16 + r + hi * 8;
            gates[(size_t)row * G4H + coln] =
                acc[r] + bhv + xprow[(size_t)row * G4H + coln];
        }

        grid_sync(bar, bar + 1, nb);

        // ---- Phase B: elementwise LSTM cell update ----
        for (int e = tid; e < B_ * H_; e += nthr) {
            const int b = e >> 9, j = e & (H_ - 1);
            const float* g = gates + (size_t)b * G4H;
            const float ig = fast_sigmoid(g[j]);
            const float fg = fast_sigmoid(g[H_ + j]);
            const float gg = fast_tanh(g[2 * H_ + j]);
            const float og = fast_sigmoid(g[3 * H_ + j]);
            const float cv = fg * c[e] + ig * gg;
            const float hv = og * fast_tanh(cv);
            c[e] = cv;
            const __hip_bfloat16 hb = __float2bfloat16(hv);
            h[e] = hb;
            if (h_seq) h_seq[(size_t)t * (B_ * H_) + e] = hb;
        }

        grid_sync(bar, bar + 1, nb);
    }
}

// ---------------- classification head: out[B,C] = h2 @ W_head + b_head ----------------
__global__ void head_kernel(const __hip_bfloat16* __restrict__ h2,
                            const float* __restrict__ W,   // [H, C]
                            const float* __restrict__ b,   // [C]
                            float* __restrict__ out) {     // [B, C]
    int bi = blockIdx.x * blockDim.x + threadIdx.x;
    if (bi >= B_) return;
    float a0 = b[0], a1 = b[1], a2 = b[2], a3 = b[3];
    for (int j = 0; j < H_; ++j) {
        const float hv = __bfloat162float(h2[(size_t)bi * H_ + j]);
        a0 += hv * W[j * 4 + 0];
        a1 += hv * W[j * 4 + 1];
        a2 += hv * W[j * 4 + 2];
        a3 += hv * W[j * 4 + 3];
    }
    out[bi * 4 + 0] = a0;
    out[bi * 4 + 1] = a1;
    out[bi * 4 + 2] = a2;
    out[bi * 4 + 3] = a3;
}

// ---------------- host launcher ----------------
extern "C" void kernel_launch(void* const* d_in, const int* in_sizes, int n_in,
                              void* d_out, int out_size, void* d_ws, size_t ws_size,
                              hipStream_t stream) {
    const float* x      = (const float*)d_in[0];
    const float* W_x1   = (const float*)d_in[1];
    const float* b_x1   = (const float*)d_in[2];
    const float* W_h1   = (const float*)d_in[3];
    const float* b_h1   = (const float*)d_in[4];
    const float* W_x2   = (const float*)d_in[5];
    const float* b_x2   = (const float*)d_in[6];
    const float* W_h2   = (const float*)d_in[7];
    const float* b_h2   = (const float*)d_in[8];
    const float* W_head = (const float*)d_in[9];
    const float* b_head = (const float*)d_in[10];
    float* out = (float*)d_out;

    char* ws = (char*)d_ws;
    size_t off = 0;
    auto take = [&](size_t bytes) -> void* {
        void* p = ws + off;
        off = (off + bytes + 255) & ~(size_t)255;
        return p;
    };

    __hip_bfloat16* WxT1  = (__hip_bfloat16*)take((size_t)G4H * I_ * 2);      // [2048,256]
    __hip_bfloat16* WhT1  = (__hip_bfloat16*)take((size_t)G4H * H_ * 2);      // [2048,512]
    __hip_bfloat16* WxT2  = (__hip_bfloat16*)take((size_t)G4H * H_ * 2);
    __hip_bfloat16* WhT2  = (__hip_bfloat16*)take((size_t)G4H * H_ * 2);
    __hip_bfloat16* xbf   = (__hip_bfloat16*)take((size_t)B_ * T_ * I_ * 2);  // 32 MB
    __hip_bfloat16* hstate= (__hip_bfloat16*)take((size_t)B_ * H_ * 2);
    float*          cstate= (float*)take((size_t)B_ * H_ * 4);
    float*          gates = (float*)take((size_t)B_ * G4H * 4);
    unsigned*       bar   = (unsigned*)take(256);
    __hip_bfloat16* h1seq = (__hip_bfloat16*)take((size_t)T_ * B_ * H_ * 2);  // 64 MB
    float*          xp    = (float*)take((size_t)T_ * B_ * G4H * 4);          // 512 MB

    const int M = T_ * B_;  // 65536 rows

    // prep: convert & transpose
    f32_to_bf16_kernel<<<(B_ * T_ * I_ + 255) / 256, 256, 0, stream>>>(x, xbf, B_ * T_ * I_);
    transpose_to_bf16_kernel<<<(I_ * G4H + 255) / 256, 256, 0, stream>>>(W_x1, WxT1, I_, G4H);
    transpose_to_bf16_kernel<<<(H_ * G4H + 255) / 256, 256, 0, stream>>>(W_h1, WhT1, H_, G4H);
    transpose_to_bf16_kernel<<<(H_ * G4H + 255) / 256, 256, 0, stream>>>(W_x2, WxT2, H_, G4H);
    transpose_to_bf16_kernel<<<(H_ * G4H + 255) / 256, 256, 0, stream>>>(W_h2, WhT2, H_, G4H);
    init_state_kernel<<<(B_ * H_ + 255) / 256, 256, 0, stream>>>(hstate, cstate, bar);

    // layer 1: xp1 = x @ W_x1 + b_x1 ; then recurrence (records h1_seq)
    {
        const int tiles = (M >> 4) * (G4H >> 4);
        wmma_gemm_bf16<<<tiles / 8, 256, 0, stream>>>(xbf, WxT1, b_x1, xp, M, G4H, I_, 1);
    }
    lstm_layer_persistent<<<128, 256, 147456, stream>>>(xp, WhT1, b_h1, hstate, cstate,
                                                        gates, h1seq, bar);

    // layer 2: reset state, xp2 = h1_seq @ W_x2 + b_x2 ; recurrence (no h_seq)
    init_state_kernel<<<(B_ * H_ + 255) / 256, 256, 0, stream>>>(hstate, cstate, bar);
    {
        const int tiles = (M >> 4) * (G4H >> 4);
        wmma_gemm_bf16<<<tiles / 8, 256, 0, stream>>>(h1seq, WxT2, b_x2, xp, M, G4H, H_, 0);
    }
    lstm_layer_persistent<<<128, 256, 147456, stream>>>(xp, WhT2, b_h2, hstate, cstate,
                                                        gates, nullptr, bar);

    // head
    head_kernel<<<1, 128, 0, stream>>>(hstate, W_head, b_head, out);
}
